// DeltaNetLayer_46892452937748
// MI455X (gfx1250) — matrix operand order repaired and checked by reading, not compiled, not run on hardware
//
#include <hip/hip_runtime.h>
#include <hip/hip_bf16.h>

// DeltaNet layer for MI455X (gfx1250, wave32, WMMA).
// Pipeline: LayerNorm -> (WMMA f16 GEMM) QKV proj -> delta-rule scan -> (WMMA f16 GEMM + residual) out proj.

typedef __attribute__((ext_vector_type(16))) _Float16 v16h;
typedef __attribute__((ext_vector_type(8)))  float    v8f;

#define Bv   4
#define Tv   2048
#define HID  1024
#define NH   16
#define Dv   64
#define Mrows (Bv * Tv)           // 8192
#define KDIM  HID                 // 1024

// ---------------------------------------------------------------------------
// LayerNorm: one block per row of 1024, write f16 normed.
// ---------------------------------------------------------------------------
__global__ __launch_bounds__(256)
void ln_kernel(const float* __restrict__ x, const float* __restrict__ g,
               const float* __restrict__ b, _Float16* __restrict__ out) {
    const int row = blockIdx.x;
    const int tid = threadIdx.x;
    const float4 v = *(const float4*)(x + (size_t)row * HID + tid * 4);

    float s  = v.x + v.y + v.z + v.w;
    float s2 = v.x * v.x + v.y * v.y + v.z * v.z + v.w * v.w;
    #pragma unroll
    for (int m = 16; m > 0; m >>= 1) {
        s  += __shfl_xor(s,  m, 32);
        s2 += __shfl_xor(s2, m, 32);
    }
    __shared__ float rs[8], rs2[8];
    if ((tid & 31) == 0) { rs[tid >> 5] = s; rs2[tid >> 5] = s2; }
    __syncthreads();
    float ts = 0.f, ts2 = 0.f;
    #pragma unroll
    for (int w = 0; w < 8; ++w) { ts += rs[w]; ts2 += rs2[w]; }
    const float mean = ts * (1.0f / HID);
    const float var  = ts2 * (1.0f / HID) - mean * mean;
    const float rstd = rsqrtf(var + 1e-5f);

    const float4 gv = *(const float4*)(g + tid * 4);
    const float4 bv = *(const float4*)(b + tid * 4);
    union { _Float16 h[4]; uint2 u; } pk;
    pk.h[0] = (_Float16)((v.x - mean) * rstd * gv.x + bv.x);
    pk.h[1] = (_Float16)((v.y - mean) * rstd * gv.y + bv.y);
    pk.h[2] = (_Float16)((v.z - mean) * rstd * gv.z + bv.z);
    pk.h[3] = (_Float16)((v.w - mean) * rstd * gv.w + bv.w);
    *(uint2*)(out + (size_t)row * HID + tid * 4) = pk.u;
}

// ---------------------------------------------------------------------------
// Transpose + f32->f16 convert: src[K][N] -> dst[N][K].
// ---------------------------------------------------------------------------
__global__ __launch_bounds__(256)
void convert_t_kernel(const float* __restrict__ src, _Float16* __restrict__ dst,
                      int K, int N) {
    const size_t idx = (size_t)blockIdx.x * 256 + threadIdx.x;
    const size_t k = idx / (size_t)N;
    const size_t n = idx - k * (size_t)N;
    dst[n * (size_t)K + k] = (_Float16)src[idx];
}

// ---------------------------------------------------------------------------
// WMMA GEMM: C[M][N] = A[M][1024] (f16, row-major) * Bt[N][1024] (f16, N-major)
//            + bias[N] (+ resid[M][N] if non-null). C is f32.
// 128x128 tile / block (256 thr = 8 waves, 2x4 wave grid).
// Each wave: 4x2 = 8 WMMA tiles of 16x16 (64x32 output per wave).
// K-loop fully unrolled (32 iters): no loop-carried acc phis -> accumulators
// stay pinned (no v_mov copies), compiler software-pipelines global->LDS->WMMA.
// ---------------------------------------------------------------------------
#define LDP 40   // padded row pitch in halves (32 data + 8 pad)

union ABfrag { uint4 u[2]; v16h h; };

__global__ __launch_bounds__(256)
void gemm_wmma_kernel(const _Float16* __restrict__ A,
                      const _Float16* __restrict__ Bt,
                      const float* __restrict__ bias,
                      const float* __restrict__ resid,   // nullable
                      float* __restrict__ C, int N) {
    __shared__ _Float16 As[128 * LDP];
    __shared__ _Float16 Bs[128 * LDP];

    const int tid = threadIdx.x;
    const int m0  = blockIdx.y * 128;
    const int n0  = blockIdx.x * 128;

    const int w   = tid >> 5;
    const int l   = tid & 31;
    const int wm  = (w & 1) * 64;   // wave row offset in tile
    const int wn  = (w >> 1) * 32;  // wave col offset in tile
    const int lm  = l & 15;
    const int lh  = l >> 4;

    // staging coords: 2 threads per row, 16 halves (two uint4) each
    const int grow = tid >> 1;
    const int gseg = (tid & 1) * 16;

    const _Float16* Ab = A  + (size_t)(m0 + grow) * KDIM + gseg;
    const _Float16* Bb = Bt + (size_t)(n0 + grow) * KDIM + gseg;

    v8f acc[4][2] = {};

    #pragma unroll
    for (int kk = 0; kk < KDIM; kk += 32) {
        __syncthreads();
        {   // stage A and B tiles (rows 0..127 of each, K cols kk..kk+31)
            const uint4* pa = (const uint4*)(Ab + kk);
            uint4 a0 = pa[0];
            uint4 a1 = pa[1];
            const uint4* pb = (const uint4*)(Bb + kk);
            uint4 b0 = pb[0];
            uint4 b1 = pb[1];
            uint4* qa = (uint4*)&As[grow * LDP + gseg];
            qa[0] = a0;
            qa[1] = a1;
            uint4* qb = (uint4*)&Bs[grow * LDP + gseg];
            qb[0] = b0;
            qb[1] = b1;
        }
        __syncthreads();

        ABfrag af[4], bf[2];
        #pragma unroll
        for (int rt = 0; rt < 4; ++rt) {
            const int r = wm + rt * 16 + lm;
            af[rt].u[0] = *(const uint4*)&As[r * LDP + 8 * lh];        // K = 8h..8h+7
            af[rt].u[1] = *(const uint4*)&As[r * LDP + 16 + 8 * lh];   // K = 16+8h..
        }
        #pragma unroll
        for (int ct = 0; ct < 2; ++ct) {
            const int r = wn + ct * 16 + lm;
            bf[ct].u[0] = *(const uint4*)&Bs[r * LDP + 8 * lh];        // K = 8h..8h+7
            bf[ct].u[1] = *(const uint4*)&Bs[r * LDP + 16 + 8 * lh];   // K = 16+8h..
        }
        #pragma unroll
        for (int rt = 0; rt < 4; ++rt)
            #pragma unroll
            for (int ct = 0; ct < 2; ++ct)
                acc[rt][ct] = __builtin_amdgcn_wmma_f32_16x16x32_f16(
                    false, af[rt].h, false, bf[ct].h,
                    (short)0, acc[rt][ct], false, false);
    }

    // epilogue: D layout -> VGPR r holds row (lh*8 + r), lanes 0..15 = cols
    #pragma unroll
    for (int ct = 0; ct < 2; ++ct) {
        const int gcol = n0 + wn + ct * 16 + lm;
        const float bz = bias[gcol];
        #pragma unroll
        for (int rt = 0; rt < 4; ++rt) {
            #pragma unroll
            for (int r = 0; r < 8; ++r) {
                const int grow2 = m0 + wm + rt * 16 + lh * 8 + r;
                float val = acc[rt][ct][r] + bz;
                if (resid) val += resid[(size_t)grow2 * HID + gcol];
                C[(size_t)grow2 * N + gcol] = val;
            }
        }
    }
}

// ---------------------------------------------------------------------------
// Delta-rule scan. One block per (b,h); 256 threads: row = tid>>2 (0..63),
// part = tid&3 owns 16 columns of S[row]. State in registers; no LDS, no
// barriers in the loop (part-reductions via wave32 shfl_xor; k/q broadcast
// loads are L0-served). Next-step operands prefetched during current compute.
// proj layout per row: [q(1024) | k(1024) | v(1024) | beta_pre(1024)].
// ---------------------------------------------------------------------------
__global__ __launch_bounds__(256)
void scan_kernel(const float* __restrict__ proj, _Float16* __restrict__ attn) {
    const int bh = blockIdx.x;
    const int b_ = bh / NH;
    const int h  = bh % NH;
    const int tid  = threadIdx.x;
    const int row  = tid >> 2;
    const int part = tid & 3;
    const int cb   = part * 16;

    const float* pbase = proj + (size_t)(b_ * Tv) * 4096 + h * Dv;

    float S[16];
    #pragma unroll
    for (int j = 0; j < 16; ++j) S[j] = 0.f;

    float4 kA[4], qA[4];
    float vA, bA;
    {   // prefetch t = 0
        const float* p = pbase;
        #pragma unroll
        for (int i = 0; i < 4; ++i) {
            qA[i] = *(const float4*)(p + cb + 4 * i);
            kA[i] = *(const float4*)(p + 1024 + cb + 4 * i);
        }
        vA = p[2048 + row];
        bA = p[3072 + row];
    }

    for (int t = 0; t < Tv; ++t) {
        float4 kB[4], qB[4];
        float vB = 0.f, bB = 0.f;
        if (t + 1 < Tv) {   // prefetch next step
            const float* p = pbase + (size_t)(t + 1) * 4096;
            #pragma unroll
            for (int i = 0; i < 4; ++i) {
                qB[i] = *(const float4*)(p + cb + 4 * i);
                kB[i] = *(const float4*)(p + 1024 + cb + 4 * i);
            }
            vB = p[2048 + row];
            bB = p[3072 + row];
        }

        // unpack + k-norm partial
        float kr[16], qr[16];
        #pragma unroll
        for (int i = 0; i < 4; ++i) {
            kr[4*i+0] = kA[i].x; kr[4*i+1] = kA[i].y; kr[4*i+2] = kA[i].z; kr[4*i+3] = kA[i].w;
            qr[4*i+0] = qA[i].x; qr[4*i+1] = qA[i].y; qr[4*i+2] = qA[i].z; qr[4*i+3] = qA[i].w;
        }
        float ss = 0.f;
        #pragma unroll
        for (int j = 0; j < 16; ++j) ss += kr[j] * kr[j];
        ss += __shfl_xor(ss, 1, 32);
        ss += __shfl_xor(ss, 2, 32);
        const float inv = 1.0f / fmaxf(sqrtf(ss), 1e-6f);
        #pragma unroll
        for (int j = 0; j < 16; ++j) kr[j] *= inv;

        // retrieval = S[row] . k
        float pr = 0.f;
        #pragma unroll
        for (int j = 0; j < 16; ++j) pr += S[j] * kr[j];
        pr += __shfl_xor(pr, 1, 32);
        pr += __shfl_xor(pr, 2, 32);

        const float beta  = 1.0f / (1.0f + expf(-bA));
        const float wcoef = beta * (vA - pr);

        // rank-1 update + output dot
        float po = 0.f;
        #pragma unroll
        for (int j = 0; j < 16; ++j) {
            S[j] += wcoef * kr[j];
            po += S[j] * qr[j];
        }
        po += __shfl_xor(po, 1, 32);
        po += __shfl_xor(po, 2, 32);

        if (part == 0)
            attn[((size_t)(b_ * Tv + t)) * HID + h * Dv + row] = (_Float16)po;

        #pragma unroll
        for (int i = 0; i < 4; ++i) { kA[i] = kB[i]; qA[i] = qB[i]; }
        vA = vB; bA = bB;
    }
}

// ---------------------------------------------------------------------------
// Launch
// ---------------------------------------------------------------------------
extern "C" void kernel_launch(void* const* d_in, const int* in_sizes, int n_in,
                              void* d_out, int out_size, void* d_ws, size_t ws_size,
                              hipStream_t stream) {
    const float* x      = (const float*)d_in[0];
    const float* ln_g   = (const float*)d_in[1];
    const float* ln_b   = (const float*)d_in[2];
    const float* w_qkvb = (const float*)d_in[3];
    const float* b_qkvb = (const float*)d_in[4];
    const float* w_out  = (const float*)d_in[5];
    const float* b_out  = (const float*)d_in[6];
    float* out = (float*)d_out;

    // workspace carve-up
    _Float16* normed = (_Float16*)d_ws;                              // 8192*1024 f16
    _Float16* wqkvT  = normed + (size_t)Mrows * HID;                 // 4096*1024 f16
    _Float16* woutT  = wqkvT + (size_t)4096 * KDIM;                  // 1024*1024 f16
    float*    proj   = (float*)(woutT + (size_t)1024 * KDIM);        // 8192*4096 f32
    _Float16* attn   = (_Float16*)(proj + (size_t)Mrows * 4096);     // 8192*1024 f16

    ln_kernel<<<Mrows, 256, 0, stream>>>(x, ln_g, ln_b, normed);

    convert_t_kernel<<<(KDIM * 4096) / 256, 256, 0, stream>>>(w_qkvb, wqkvT, KDIM, 4096);
    convert_t_kernel<<<(KDIM * 1024) / 256, 256, 0, stream>>>(w_out,  woutT, KDIM, 1024);

    // QKV projection: [8192,1024] x [1024,4096]
    gemm_wmma_kernel<<<dim3(4096 / 128, Mrows / 128), 256, 0, stream>>>(
        normed, wqkvT, b_qkvb, nullptr, proj, 4096);

    // delta-rule scan over T
    scan_kernel<<<Bv * NH, 256, 0, stream>>>(proj, attn);

    // output projection + residual: out = x + attn @ w_out + b_out
    gemm_wmma_kernel<<<dim3(1024 / 128, Mrows / 128), 256, 0, stream>>>(
        attn, woutT, b_out, x, out, 1024);
}
